// AdaptiveBlockSelfAttention_10857677325158
// MI455X (gfx1250) — matrix operand order repaired and checked by reading, not compiled, not run on hardware
//
#include <hip/hip_runtime.h>

// ---------------------------------------------------------------------------
// AdaptiveBlockSelfAttention for MI455X (gfx1250, wave32, WMMA bf16)
// B=4, C=192, H=W=256, BLOCK=16 -> N=P=256, hidden=384
// ---------------------------------------------------------------------------

typedef __attribute__((ext_vector_type(16))) __bf16 v16bf;
typedef __attribute__((ext_vector_type(8)))  __bf16 bf16x8;
typedef __attribute__((ext_vector_type(8)))  float  v8f;

#define DEV static __device__ __forceinline__

// A-operand fragment (16x32 bf16). Row-major source, ld in elements.
// lane m = lane&15 selects M row; elems 0..7 hold K = h*8+0..7, elems 8..15
// hold K = 16+h*8+0..7 (h = lane>>4), matching the CDNA5 16-bit A layout.
DEV v16bf load_frag_a(const __bf16* base, int ld, int row0, int k0, int lane) {
  const int m = lane & 15, hf = lane >> 4;
  const __bf16* p = base + (size_t)(row0 + m) * ld + (k0 + hf * 8);
  bf16x8 lo = *(const bf16x8*)(p);
  bf16x8 hi = *(const bf16x8*)(p + 16);
  v16bf r;
#pragma unroll
  for (int i = 0; i < 8; ++i) { r[i] = lo[i]; r[i + 8] = hi[i]; }
  return r;
}

// B-operand fragment (32x16 bf16) built from a transposed (NxK row-major)
// source: lane n = lane&15 selects the N column, elems 0..7 hold K = h*8+0..7,
// elems 8..15 hold K = 16+h*8+0..7 (h = lane>>4), matching the A layout.
DEV v16bf load_frag_b(const __bf16* base, int ld, int n0, int k0, int lane) {
  const int n = lane & 15, kb = (lane >> 4) * 8;
  const __bf16* p = base + (size_t)(n0 + n) * ld + (k0 + kb);
  bf16x8 lo = *(const bf16x8*)(p);
  bf16x8 hi = *(const bf16x8*)(p + 16);
  v16bf r;
#pragma unroll
  for (int i = 0; i < 8; ++i) { r[i] = lo[i]; r[i + 8] = hi[i]; }
  return r;
}

DEV v8f wmma_bf16(v16bf a, v16bf b, v8f c) {
  return __builtin_amdgcn_wmma_f32_16x16x32_bf16(false, a, false, b,
                                                 (short)0, c, false, false);
}

DEV v8f zero8() { v8f z = {0.f, 0.f, 0.f, 0.f, 0.f, 0.f, 0.f, 0.f}; return z; }

// ---------------------------------------------------------------------------
// Kernel 0a: weights f32 -> bf16 (row-major, K contiguous = A-operand layout)
// ---------------------------------------------------------------------------
__global__ __launch_bounds__(256) void k_convert_weights(
    const float* __restrict__ wq, const float* __restrict__ wk,
    const float* __restrict__ wv, const float* __restrict__ wf1,
    const float* __restrict__ wf2, __bf16* __restrict__ dst) {
  const int i = blockIdx.x * 256 + threadIdx.x;
  if (i < 36864)        dst[i] = (__bf16)wq[i];
  else if (i < 73728)   dst[i] = (__bf16)wk[i - 36864];
  else if (i < 110592)  dst[i] = (__bf16)wv[i - 73728];
  else if (i < 184320)  dst[i] = (__bf16)wf1[i - 110592];
  else if (i < 258048)  dst[i] = (__bf16)wf2[i - 184320];
}

// ---------------------------------------------------------------------------
// Kernel 0b: pack x (B,C,H,W f32) -> xb (B, s, C) bf16 with s = p*256 + n
//   n = (h>>4)*16 + (w>>4)  (block index), p = (h&15)*16 + (w&15)
// LDS-tiled so both global read (along w) and write (along c) are coalesced.
// ---------------------------------------------------------------------------
__global__ __launch_bounds__(256) void k_pack_x(const float* __restrict__ x,
                                                __bf16* __restrict__ xb) {
  int gid = blockIdx.x;
  const int ct = gid % 6;  gid /= 6;     // channel tile (32)
  const int wt = gid % 4;  gid /= 4;     // w tile (64)
  const int h  = gid % 256; gid /= 256;
  const int b  = gid;
  const int c0 = ct * 32, w0 = wt * 64;
  __shared__ float tile[32][68];
  const int t = threadIdx.x;
  const int tx = t & 63, ty = t >> 6;    // ty: 0..3
#pragma unroll
  for (int j = 0; j < 8; ++j) {
    const int cl = ty + 4 * j;           // 0..31
    tile[cl][tx] =
        x[(((size_t)b * 192 + (c0 + cl)) * 256 + h) * 256 + (w0 + tx)];
  }
  __syncthreads();
  const int px = t >> 2, cp = t & 3;     // 64 pixels x 4 chan-chunks
  const int w = w0 + px;
  const int n = ((h >> 4) << 4) + (w >> 4);
  const int p = ((h & 15) << 4) + (w & 15);
  const int s = p * 256 + n;
  __bf16* dst = xb + ((size_t)b * 65536 + s) * 192 + c0 + cp * 8;
#pragma unroll
  for (int j = 0; j < 8; ++j) dst[j] = (__bf16)tile[cp * 8 + j][px];
}

// ---------------------------------------------------------------------------
// Kernel 1: QKV projection. GEMM: out[c, s] = sum_cin W[c,cin] * xb[s,cin].
// Workgroup = 8 waves = 64 out-channels x 32 pixels; each wave computes a
// 16x16 tile for all of Q, K, V (shared B fragment). Outputs land directly in
// the transposed attention layout Qt/Kt/Vt[(b,c)][p*256 + n].
// ---------------------------------------------------------------------------
__global__ __launch_bounds__(256) void k_qkv(
    const __bf16* __restrict__ xb,
    const __bf16* __restrict__ wqb, const __bf16* __restrict__ wkb,
    const __bf16* __restrict__ wvb,
    const float* __restrict__ bq, const float* __restrict__ bk,
    const float* __restrict__ bv,
    __bf16* __restrict__ qt, __bf16* __restrict__ kt,
    __bf16* __restrict__ vt) {
  int gid = blockIdx.x;
  const int ctile = gid % 3;    gid /= 3;
  const int st    = gid % 2048; gid /= 2048;
  const int b     = gid;
  const int lane = threadIdx.x & 31, wave = threadIdx.x >> 5;
  const int c0 = ctile * 64 + (wave >> 1) * 16;
  const int s0 = st * 32 + (wave & 1) * 16;
  const __bf16* xbb = xb + (size_t)b * 65536 * 192;
  v8f aq = zero8(), ak = zero8(), av = zero8();
#pragma unroll
  for (int kk = 0; kk < 6; ++kk) {
    const int k0 = kk * 32;
    v16bf bf = load_frag_b(xbb, 192, s0, k0, lane);
    aq = wmma_bf16(load_frag_a(wqb, 192, c0, k0, lane), bf, aq);
    ak = wmma_bf16(load_frag_a(wkb, 192, c0, k0, lane), bf, ak);
    av = wmma_bf16(load_frag_a(wvb, 192, c0, k0, lane), bf, av);
  }
  const int hf = lane >> 4, ln = lane & 15;
  const int s_out = s0 + ln;
#pragma unroll
  for (int r = 0; r < 8; ++r) {
    const int c = c0 + r + hf * 8;               // D: M = r + 8*(lane>=16)
    const size_t o = ((size_t)b * 192 + c) * 65536 + s_out;
    qt[o] = (__bf16)(aq[r] + bq[c]);
    kt[o] = (__bf16)(ak[r] + bk[c]);
    vt[o] = (__bf16)(av[r] + bv[c]);
  }
}

// ---------------------------------------------------------------------------
// Kernel 2: S = Qt * Kt^T / sqrt(C), row softmax, attn -> bf16.
// One workgroup per (b, c, 16-row p-tile); each wave owns key tiles w and w+8.
// ---------------------------------------------------------------------------
__global__ __launch_bounds__(256) void k_scores_softmax(
    const __bf16* __restrict__ qt, const __bf16* __restrict__ kt,
    __bf16* __restrict__ attn) {
  int gid = blockIdx.x;
  const int pt = gid & 15; gid >>= 4;
  const int c  = gid % 192; gid /= 192;
  const int b  = gid;
  const int p0 = pt * 16;
  const size_t mo = ((size_t)b * 192 + c) * 65536;
  const __bf16* qb = qt + mo;
  const __bf16* kb = kt + mo;
  const int lane = threadIdx.x & 31, wave = threadIdx.x >> 5;
  v8f a0 = zero8(), a1 = zero8();
#pragma unroll
  for (int kk = 0; kk < 8; ++kk) {
    const int k0 = kk * 32;
    v16bf af = load_frag_a(qb, 256, p0, k0, lane);
    a0 = wmma_bf16(af, load_frag_b(kb, 256, wave * 16, k0, lane), a0);
    a1 = wmma_bf16(af, load_frag_b(kb, 256, (wave + 8) * 16, k0, lane), a1);
  }
  __shared__ float S[16][256];
  __shared__ float red[16][17];
  const float scale = 0.0721687836487032206f;    // 1/sqrt(192)
  const int hf = lane >> 4, ln = lane & 15;
#pragma unroll
  for (int r = 0; r < 8; ++r) {
    S[r + hf * 8][wave * 16 + ln]       = a0[r] * scale;
    S[r + hf * 8][(wave + 8) * 16 + ln] = a1[r] * scale;
  }
  __syncthreads();
  const int row = threadIdx.x >> 4, seg = threadIdx.x & 15;
  float mx = -3.40282347e38f;
#pragma unroll
  for (int j = 0; j < 16; ++j) mx = fmaxf(mx, S[row][seg * 16 + j]);
  red[row][seg] = mx;
  __syncthreads();
  for (int stp = 8; stp >= 1; stp >>= 1) {
    if (seg < stp) red[row][seg] = fmaxf(red[row][seg], red[row][seg + stp]);
    __syncthreads();
  }
  const float rmax = red[row][0];
  __syncthreads();
  float sum = 0.f;
#pragma unroll
  for (int j = 0; j < 16; ++j) {
    const float e = __expf(S[row][seg * 16 + j] - rmax);
    S[row][seg * 16 + j] = e;
    sum += e;
  }
  red[row][seg] = sum;
  __syncthreads();
  for (int stp = 8; stp >= 1; stp >>= 1) {
    if (seg < stp) red[row][seg] += red[row][seg + stp];
    __syncthreads();
  }
  const float inv = 1.0f / red[row][0];
  __bf16* ab = attn + mo + (size_t)(p0 + row) * 256 + seg * 16;
#pragma unroll
  for (int j = 0; j < 16; ++j) ab[j] = (__bf16)(S[row][seg * 16 + j] * inv);
}

// ---------------------------------------------------------------------------
// Kernel 3: O = attn * V, fused residual + (p,l)->(h,w) un-blocking.
// Writes xs (f32, BCHW, for later residuals) and xst (bf16, pixel-major,
// channel-contiguous: B-operand of the FFN).
// ---------------------------------------------------------------------------
__global__ __launch_bounds__(256) void k_attn_out(
    const __bf16* __restrict__ attn, const __bf16* __restrict__ vt,
    const float* __restrict__ x, float* __restrict__ xs,
    __bf16* __restrict__ xst) {
  int gid = blockIdx.x;
  const int pt = gid & 15; gid >>= 4;
  const int c  = gid % 192; gid /= 192;
  const int b  = gid;
  const int p0 = pt * 16;
  const size_t mo = ((size_t)b * 192 + c) * 65536;
  const __bf16* ab = attn + mo;
  const __bf16* vb = vt + mo;
  const int lane = threadIdx.x & 31, wave = threadIdx.x >> 5;
  v8f o0 = zero8(), o1 = zero8();
#pragma unroll
  for (int kk = 0; kk < 8; ++kk) {
    const int k0 = kk * 32;
    v16bf af = load_frag_a(ab, 256, p0, k0, lane);
    o0 = wmma_bf16(af, load_frag_b(vb, 256, wave * 16, k0, lane), o0);
    o1 = wmma_bf16(af, load_frag_b(vb, 256, (wave + 8) * 16, k0, lane), o1);
  }
  const int hf = lane >> 4, ln = lane & 15;
#pragma unroll
  for (int r = 0; r < 8; ++r) {
    const int p = p0 + r + hf * 8;
#pragma unroll
    for (int half = 0; half < 2; ++half) {
      const int l = (wave + half * 8) * 16 + ln;
      const float o = half ? o1[r] : o0[r];
      const int hh  = ((p >> 4) << 4) + (l >> 4);
      const int wwp = ((p & 15) << 4) + (l & 15);
      const int hw = hh * 256 + wwp;
      const float v = x[mo + hw] + o;
      xs[mo + hw] = v;
      xst[((size_t)b * 65536 + hw) * 192 + c] = (__bf16)v;
    }
  }
}

// ---------------------------------------------------------------------------
// Kernel 4: fused FFN. h = gelu(Wf1*xs + bf1) staged transposed in LDS bf16,
// y = Wf2*h + bf2; out = xs + y. One workgroup per 32-pixel tile.
// ---------------------------------------------------------------------------
__global__ __launch_bounds__(256) void k_ffn(
    const __bf16* __restrict__ xst, const __bf16* __restrict__ wf1b,
    const __bf16* __restrict__ wf2b, const float* __restrict__ bf1,
    const float* __restrict__ bf2, const float* __restrict__ xs,
    float* __restrict__ out) {
  int gid = blockIdx.x;
  const int pxt = gid % 2048; gid /= 2048;
  const int b = gid;
  const int pix0 = pxt * 32;
  const int lane = threadIdx.x & 31, wave = threadIdx.x >> 5;
  const int hf = lane >> 4, ln = lane & 15;
  __shared__ __bf16 hbuf[32][384];               // [pixel][hidden], 24 KB
  const __bf16* xr = xst + (size_t)b * 65536 * 192;
  // --- GEMM1: hidden(384) x pixels(32), K = 192 ---
  for (int tt = wave; tt < 48; tt += 8) {        // 24 h-tiles x 2 px-tiles
    const int ht = tt % 24, pxh = tt / 24;
    const int h0 = ht * 16;
    v8f acc = zero8();
#pragma unroll
    for (int kk = 0; kk < 6; ++kk) {
      const int k0 = kk * 32;
      acc = wmma_bf16(load_frag_a(wf1b, 192, h0, k0, lane),
                      load_frag_b(xr, 192, pix0 + pxh * 16, k0, lane), acc);
    }
    const int pc = pxh * 16 + ln;
#pragma unroll
    for (int r = 0; r < 8; ++r) {
      const int hr = h0 + r + hf * 8;
      const float v = acc[r] + bf1[hr];
      const float g = 0.5f * v * (1.0f + erff(v * 0.70710678118654752f));
      hbuf[pc][hr] = (__bf16)g;                  // store transposed for GEMM2
    }
  }
  __syncthreads();
  // --- GEMM2: C(192) x pixels(32), K = 384, B-operand from LDS ---
  for (int tt = wave; tt < 24; tt += 8) {        // 12 c-tiles x 2 px-tiles
    const int ct = tt % 12, pxh = tt / 12;
    const int c0 = ct * 16;
    v8f acc = zero8();
#pragma unroll
    for (int kk = 0; kk < 12; ++kk) {
      const int k0 = kk * 32;
      acc = wmma_bf16(load_frag_a(wf2b, 384, c0, k0, lane),
                      load_frag_b(&hbuf[0][0], 384, pxh * 16, k0, lane), acc);
    }
    const int hw = pix0 + pxh * 16 + ln;
#pragma unroll
    for (int r = 0; r < 8; ++r) {
      const int c = c0 + r + hf * 8;
      const size_t idx = ((size_t)b * 192 + c) * 65536 + hw;
      out[idx] = xs[idx] + acc[r] + bf2[c];
    }
  }
}

// ---------------------------------------------------------------------------
extern "C" void kernel_launch(void* const* d_in, const int* in_sizes, int n_in,
                              void* d_out, int out_size, void* d_ws,
                              size_t ws_size, hipStream_t stream) {
  const float* x   = (const float*)d_in[0];
  const float* Wq  = (const float*)d_in[1];
  const float* bq  = (const float*)d_in[2];
  const float* Wk  = (const float*)d_in[3];
  const float* bk  = (const float*)d_in[4];
  const float* Wv  = (const float*)d_in[5];
  const float* bv  = (const float*)d_in[6];
  const float* Wf1 = (const float*)d_in[7];
  const float* bf1 = (const float*)d_in[8];
  const float* Wf2 = (const float*)d_in[9];
  const float* bf2 = (const float*)d_in[10];
  float* out = (float*)d_out;

  // ---- workspace layout (bytes) ----
  // [0, 1MB)           : bf16 weights (wq,wk,wv,wf1,wf2)
  // [1MB, +96MB)       : xb (pack of x)      -> reused as attn after k_qkv
  // next 96MB          : Qt  -> reused (with Kt) as xs f32 after k_scores
  // next 96MB          : Kt
  // next 96MB          : Vt
  // next 96MB          : xst (bf16 pixel-major x_sa)
  char* ws = (char*)d_ws;
  __bf16* wqb  = (__bf16*)ws;
  __bf16* wkb  = wqb + 36864;
  __bf16* wvb  = wkb + 36864;
  __bf16* wf1b = wvb + 36864;
  __bf16* wf2b = wf1b + 73728;
  const size_t NE = (size_t)4 * 65536 * 192;     // 50,331,648 elements
  __bf16* xb   = (__bf16*)(ws + (size_t)(1u << 20));
  __bf16* qt   = xb + NE;
  __bf16* kt   = qt + NE;
  __bf16* vt   = kt + NE;
  __bf16* xst  = vt + NE;
  __bf16* attn = xb;           // xb dead after k_qkv
  float*  xs   = (float*)qt;   // qt+kt dead after k_scores_softmax

  k_convert_weights<<<1008, 256, 0, stream>>>(Wq, Wk, Wv, Wf1, Wf2, wqb);
  k_pack_x<<<4 * 256 * 4 * 6, 256, 0, stream>>>(x, xb);           // 24576
  k_qkv<<<4 * 2048 * 3, 256, 0, stream>>>(xb, wqb, wkb, wvb, bq, bk, bv,
                                          qt, kt, vt);            // 24576
  k_scores_softmax<<<4 * 192 * 16, 256, 0, stream>>>(qt, kt, attn); // 12288
  k_attn_out<<<4 * 192 * 16, 256, 0, stream>>>(attn, vt, x, xs, xst);
  k_ffn<<<4 * 2048, 256, 0, stream>>>(xst, wf1b, wf2b, bf1, bf2, xs, out);
}
